// GlobalLayer_3521873183275
// MI455X (gfx1250) — matrix operand order repaired and checked by reading, not compiled, not run on hardware
//
#include <hip/hip_runtime.h>
#include <hip/hip_bf16.h>
#include <stdint.h>

// Problem constants (reference: B=4, N=2048, F_IN=F_OUT=30, H=6, D=5, TEMP=0.3)
#define BB    4
#define NN    2048
#define FIN   30
#define FPAD  32
#define KEEP  342            // N - (N-1 - N//6) = 2048 - 1706
#define NEG_FILL (-1e-07f)
#define INV_TEMP (1.0f/0.3f)
#define QSCALE 0.44721359549995793f   // D^-0.5 = 1/sqrt(5)

typedef __attribute__((ext_vector_type(16))) _Float16 v16h;
typedef __attribute__((ext_vector_type(8)))  _Float16 v8h;
typedef __attribute__((ext_vector_type(8)))  float    v8f;

__device__ __forceinline__ uint32_t fkey(float f) {
  // order-preserving float->uint map (ascending)
  uint32_t u = __float_as_uint(f);
  return (u & 0x80000000u) ? ~u : (u | 0x80000000u);
}

// ---------------------------------------------------------------------------
// Kernel 1: fused embed + leaky_relu + QK projection; emit f16 Q (scaled by
// D^-0.5), f16 K, and f16 h-transposed (hT[b][f][n]) padded 30->32.
// ---------------------------------------------------------------------------
__global__ __launch_bounds__(256) void proj_qk_kernel(
    const float* __restrict__ h, const float* __restrict__ coor,
    const float* __restrict__ W_embed, const float* __restrict__ b_embed,
    const float* __restrict__ W_qk, const float* __restrict__ b_qk,
    _Float16* __restrict__ Qh, _Float16* __restrict__ Kh,
    _Float16* __restrict__ hT)
{
  int idx = blockIdx.x * blockDim.x + threadIdx.x;   // row id in [0, B*N)
  if (idx >= BB * NN) return;
  int b = idx / NN, n = idx % NN;

  const float* hrow = h + (size_t)idx * FIN;
  const float* crow = coor + (size_t)idx * 3;
  float c0 = crow[0], c1 = crow[1], c2 = crow[2];

  float he[FIN];
  for (int j = 0; j < FIN; ++j) {
    float e = c0 * W_embed[j] + c1 * W_embed[FIN + j] + c2 * W_embed[2*FIN + j]
            + b_embed[j];
    e = (e > 0.0f) ? e : 0.1f * e;           // leaky_relu slope 0.1
    he[j] = hrow[j] + e;
  }

  _Float16* qrow = Qh + (size_t)idx * FPAD;
  _Float16* krow = Kh + (size_t)idx * FPAD;
  // qk[..., h, d, p]: q = qk[2j], k = qk[2j+1] with j = h*D + d
  for (int j = 0; j < FIN; ++j) {
    float q = b_qk[2*j], k = b_qk[2*j + 1];
    #pragma unroll
    for (int f = 0; f < FIN; ++f) {
      q += he[f] * W_qk[f * 2*FIN + 2*j];
      k += he[f] * W_qk[f * 2*FIN + 2*j + 1];
    }
    qrow[j] = (_Float16)(q * QSCALE);
    krow[j] = (_Float16)k;
  }
  qrow[30] = qrow[31] = (_Float16)0.0f;
  krow[30] = krow[31] = (_Float16)0.0f;

  // value matrix = ORIGINAL h, stored transposed: hT[(b*32+f)*N + n]
  for (int f = 0; f < FIN; ++f)
    hT[((size_t)b * FPAD + f) * NN + n] = (_Float16)hrow[f];
  hT[((size_t)b * FPAD + 30) * NN + n] = (_Float16)0.0f;
  hT[((size_t)b * FPAD + 31) * NN + n] = (_Float16)0.0f;
}

// ---------------------------------------------------------------------------
// Kernel 2: one workgroup per (batch, 16-query tile).
//  Stage 1: scores S = Q*K^T via v_wmma_f32_16x16x32_f16 -> LDS [16][2048] f32
//  Stage 2: exact 342nd-largest threshold (bitwise binary search on keys),
//           mask -> -1e-7, softmax(x/0.3), probs overwrite LDS in place
//  Stage 3: out = P @ h via WMMA (probs f32->f16 in regs), 8-wave K-split
//  Stage 4: out @ W_out + b_out (30x30, VALU)
// ---------------------------------------------------------------------------
#define SC_BYTES   (16 * NN * 4)                 // 131072
#define PART_BYTES (8 * 2 * 32 * 8 * 4)          // 16384
#define OUTF_BYTES (16 * FPAD * 4)               // 2048
#define SMEM_BYTES (SC_BYTES + PART_BYTES + OUTF_BYTES)

__global__ __launch_bounds__(256) void attn_wmma_kernel(
    const _Float16* __restrict__ Qh, const _Float16* __restrict__ Kh,
    const _Float16* __restrict__ hT,
    const float* __restrict__ W_out, const float* __restrict__ b_out,
    float* __restrict__ out)
{
  extern __shared__ char smem_raw[];
  float* sc    = (float*)smem_raw;                       // [16][NN]
  float* parts = (float*)(smem_raw + SC_BYTES);          // [8][2][32][8]
  float* outf  = (float*)(smem_raw + SC_BYTES + PART_BYTES); // [16][32]

  const int tid  = threadIdx.x;
  const int wave = tid >> 5;
  const int lane = tid & 31;
  const int b    = blockIdx.x / (NN / 16);
  const int qt   = blockIdx.x % (NN / 16);
  const int q0   = qt * 16;

  const int mrow = lane & 15;
  const int abase = (lane < 16) ? 0 : 8;    // unused; A fragment also uses kbase
  const int kbase = (lane < 16) ? 0 : 16;   // B-fragment K base (halves)

  // ---- Stage 1: score tiles --------------------------------------------
  // A fragment (shared by every key tile): row M = lane&15 of the Q tile.
  const _Float16* qrow = Qh + ((size_t)(b * NN + q0 + mrow)) * FPAD;
  v8h qlo = *(const v8h*)(qrow + kbase);
  v8h qhi = *(const v8h*)(qrow + kbase + 8);
  v16h afrag;
  #pragma unroll
  for (int i = 0; i < 8; ++i) { afrag[i] = qlo[i]; afrag[i + 8] = qhi[i]; }

  for (int jt = wave; jt < NN / 16; jt += 8) {
    const _Float16* krow =
        Kh + ((size_t)(b * NN + jt * 16 + (lane & 15))) * FPAD + kbase;
    if (jt + 8 < NN / 16)
      __builtin_prefetch(krow + 8 * 16 * FPAD, 0, 0);   // global_prefetch_b8
    v16h bfrag = *(const v16h*)krow;
    v8f c = {};
    v8f d = __builtin_amdgcn_wmma_f32_16x16x32_f16(
        false, afrag, false, bfrag, (short)0, c, false, false);
    #pragma unroll
    for (int r = 0; r < 8; ++r) {
      int row = r + ((lane < 16) ? 0 : 8);
      sc[row * NN + jt * 16 + (lane & 15)] = d[r];
    }
  }
  __syncthreads();

  // ---- Stage 2: threshold + softmax, one wave per pair of rows ----------
  for (int rr = 0; rr < 2; ++rr) {
    int row = wave * 2 + rr;
    float* rowp = sc + row * NN;

    // exact 342nd-largest: max key T with count(key >= T) >= KEEP
    uint32_t thr = 0u;
    for (int bit = 31; bit >= 0; --bit) {
      uint32_t cand = thr | (1u << bit);
      int cnt = 0;
      for (int e = lane; e < NN; e += 32)
        cnt += (fkey(rowp[e]) >= cand) ? 1 : 0;
      #pragma unroll
      for (int o = 16; o > 0; o >>= 1) cnt += __shfl_xor(cnt, o, 32);
      if (cnt >= KEEP) thr = cand;
    }

    // masked row max
    float m = -3.402823e38f;
    for (int e = lane; e < NN; e += 32) {
      float v = rowp[e];
      float vv = (fkey(v) >= thr) ? v : NEG_FILL;
      m = fmaxf(m, vv);
    }
    #pragma unroll
    for (int o = 16; o > 0; o >>= 1) m = fmaxf(m, __shfl_xor(m, o, 32));

    // exp((x-m)/T), write back, sum
    float s = 0.0f;
    for (int e = lane; e < NN; e += 32) {
      float v = rowp[e];
      float vv = (fkey(v) >= thr) ? v : NEG_FILL;
      float p = __expf((vv - m) * INV_TEMP);
      rowp[e] = p;
      s += p;
    }
    #pragma unroll
    for (int o = 16; o > 0; o >>= 1) s += __shfl_xor(s, o, 32);

    float inv = 1.0f / s;
    for (int e = lane; e < NN; e += 32) rowp[e] *= inv;
  }
  __syncthreads();

  // ---- Stage 3: out_feat = P @ h  (two 16-col tiles, 8-wave K split) -----
  v8f acc0 = {}, acc1 = {};
  for (int jt = wave; jt < NN / 32; jt += 8) {
    // A fragment: probs row M, K-chunk jt*32..+31, f32->f16 in regs
    const float* prow = sc + mrow * NN + jt * 32;
    v16h pa;
    #pragma unroll
    for (int i = 0; i < 8; ++i) pa[i] = (_Float16)prow[kbase + i];
    #pragma unroll
    for (int i = 0; i < 8; ++i) pa[i + 8] = (_Float16)prow[kbase + 8 + i];

    const _Float16* h0 =
        hT + ((size_t)b * FPAD + (lane & 15)) * NN + jt * 32 + kbase;
    const _Float16* h1 =
        hT + ((size_t)b * FPAD + 16 + (lane & 15)) * NN + jt * 32 + kbase;
    v16h b0 = *(const v16h*)h0;
    v16h b1 = *(const v16h*)h1;
    acc0 = __builtin_amdgcn_wmma_f32_16x16x32_f16(
        false, pa, false, b0, (short)0, acc0, false, false);
    acc1 = __builtin_amdgcn_wmma_f32_16x16x32_f16(
        false, pa, false, b1, (short)0, acc1, false, false);
  }
  #pragma unroll
  for (int r = 0; r < 8; ++r) {
    parts[((wave * 2 + 0) * 32 + lane) * 8 + r] = acc0[r];
    parts[((wave * 2 + 1) * 32 + lane) * 8 + r] = acc1[r];
  }
  __syncthreads();

  if (wave < 2) {               // wave t reduces output-column tile t
    int t = wave;
    #pragma unroll
    for (int r = 0; r < 8; ++r) {
      float f = 0.0f;
      #pragma unroll
      for (int w = 0; w < 8; ++w)
        f += parts[((w * 2 + t) * 32 + lane) * 8 + r];
      int row = r + ((lane < 16) ? 0 : 8);
      int col = t * 16 + (lane & 15);
      outf[row * FPAD + col] = f;
    }
  }
  __syncthreads();

  // ---- Stage 4: final 30x30 projection ----------------------------------
  for (int idx = tid; idx < 16 * FIN; idx += 256) {
    int row = idx / FIN, o = idx % FIN;
    float v = b_out[o];
    #pragma unroll
    for (int f = 0; f < FIN; ++f) v += outf[row * FPAD + f] * W_out[f * FIN + o];
    out[((size_t)b * NN + q0 + row) * FIN + o] = v;
  }
}

// ---------------------------------------------------------------------------
extern "C" void kernel_launch(void* const* d_in, const int* in_sizes, int n_in,
                              void* d_out, int out_size, void* d_ws, size_t ws_size,
                              hipStream_t stream) {
  const float* h       = (const float*)d_in[0];
  const float* coor    = (const float*)d_in[1];
  const float* W_embed = (const float*)d_in[2];
  const float* b_embed = (const float*)d_in[3];
  const float* W_qk    = (const float*)d_in[4];
  const float* b_qk    = (const float*)d_in[5];
  const float* W_out   = (const float*)d_in[6];
  const float* b_out   = (const float*)d_in[7];
  float* out = (float*)d_out;

  // workspace: Qh [B*N][32] f16, Kh [B*N][32] f16, hT [B][32][N] f16
  _Float16* Qh = (_Float16*)d_ws;                          // 512 KB
  _Float16* Kh = Qh + (size_t)BB * NN * FPAD;              // 512 KB
  _Float16* hT = Kh + (size_t)BB * NN * FPAD;              // 512 KB

  proj_qk_kernel<<<dim3((BB * NN + 255) / 256), dim3(256), 0, stream>>>(
      h, coor, W_embed, b_embed, W_qk, b_qk, Qh, Kh, hT);

  attn_wmma_kernel<<<dim3(BB * (NN / 16)), dim3(256), SMEM_BYTES, stream>>>(
      Qh, Kh, hT, W_out, b_out, out);
}
